// D_point_AE_fc_33706903339267
// MI455X (gfx1250) — matrix operand order repaired and checked by reading, not compiled, not run on hardware
//
#include <hip/hip_runtime.h>
#include <hip/hip_fp16.h>

typedef _Float16 half_t;
typedef __attribute__((ext_vector_type(16))) _Float16 v16h;
typedef __attribute__((ext_vector_type(8)))  _Float16 v8h;
typedef __attribute__((ext_vector_type(8)))  float    v8f;

static constexpr int WPB    = 8;    // waves per block in the GEMM
static constexpr int NCHUNK = 64;   // partial-sum chunks for BN stats
static constexpr int NB  = 32;      // batch
static constexpr int NPT = 2048;    // points
static constexpr int S1v = 512;
static constexpr int S2v = 128;
static constexpr int NSv = 64;      // nsample

// ---------------------------------------------------------------------------
// Weight convert: f32 (cout x cin) -> f16 padded (coutPad x cinPad), zero-fill
// ---------------------------------------------------------------------------
__global__ void k_convert_w(const float* __restrict__ w, half_t* __restrict__ wp,
                            int cout, int cin, int cinPad, long total) {
  long t = (long)blockIdx.x * blockDim.x + threadIdx.x;
  if (t >= total) return;
  int r = (int)(t / cinPad), c = (int)(t % cinPad);
  wp[t] = (r < cout && c < cin) ? (half_t)w[(size_t)r * cin + c] : (half_t)0.0f;
}

// ---------------------------------------------------------------------------
// Farthest point sampling: one workgroup per batch, coords + dist in LDS
// ---------------------------------------------------------------------------
__global__ __launch_bounds__(256)
void k_fps(const float* __restrict__ pts, long bStride, long pStride, long cStride,
           int n, int npoint, int* __restrict__ idxOut, float* __restrict__ newXyz) {
  __shared__ float lx[NPT], ly[NPT], lz[NPT], dist[NPT];
  __shared__ float rv[256];
  __shared__ int   ri[256];
  __shared__ int   s_far;
  int b = blockIdx.x, tid = threadIdx.x;
  const float* base = pts + (size_t)b * bStride;
  for (int p = tid; p < n; p += 256) {
    lx[p] = base[(size_t)p * pStride + 0 * cStride];
    ly[p] = base[(size_t)p * pStride + 1 * cStride];
    lz[p] = base[(size_t)p * pStride + 2 * cStride];
    dist[p] = 1e10f;
  }
  if (tid == 0) s_far = 0;
  __syncthreads();
  for (int it = 0; it < npoint; ++it) {
    int far = s_far;
    float cx = lx[far], cy = ly[far], cz = lz[far];
    if (tid == 0) {
      idxOut[b * npoint + it] = far;
      float* nx = newXyz + (size_t)(b * npoint + it) * 3;
      nx[0] = cx; nx[1] = cy; nx[2] = cz;
    }
    float bestv = -1.0f; int besti = 0;
    for (int p = tid; p < n; p += 256) {
      float dx = lx[p] - cx, dy = ly[p] - cy, dz = lz[p] - cz;
      float d = dx * dx + dy * dy + dz * dz;
      float m = fminf(dist[p], d);
      dist[p] = m;
      if (m > bestv) { bestv = m; besti = p; }
    }
    rv[tid] = bestv; ri[tid] = besti;
    __syncthreads();
    for (int s = 128; s > 0; s >>= 1) {
      if (tid < s && rv[tid + s] > rv[tid]) { rv[tid] = rv[tid + s]; ri[tid] = ri[tid + s]; }
      __syncthreads();
    }
    if (tid == 0) s_far = ri[0];
    __syncthreads();
  }
}

// ---------------------------------------------------------------------------
// Ball query: one thread per center; in-order scan == sorted-index semantics
// ---------------------------------------------------------------------------
__global__ void k_ball(const float* __restrict__ pts, long bStride, long pStride, long cStride,
                       int n, const float* __restrict__ newXyz, int S, float r2,
                       int* __restrict__ idxOut, int total) {
  int t = blockIdx.x * blockDim.x + threadIdx.x;
  if (t >= total) return;
  int b = t / S;
  const float* base = pts + (size_t)b * bStride;
  float cx = newXyz[(size_t)t * 3 + 0];
  float cy = newXyz[(size_t)t * 3 + 1];
  float cz = newXyz[(size_t)t * 3 + 2];
  int* o = idxOut + (size_t)t * NSv;
  int cnt = 0, first = 0;
  for (int p = 0; p < n && cnt < NSv; ++p) {
    float dx = base[(size_t)p * pStride + 0 * cStride] - cx;
    float dy = base[(size_t)p * pStride + 1 * cStride] - cy;
    float dz = base[(size_t)p * pStride + 2 * cStride] - cz;
    float d = dx * dx + dy * dy + dz * dz;
    if (d <= r2) { if (cnt == 0) first = p; o[cnt++] = p; }
  }
  for (int j = cnt; j < NSv; ++j) o[j] = first;
}

// ---------------------------------------------------------------------------
// Grouping kernels: gather + center + concat -> f16 padded activations
// ---------------------------------------------------------------------------
__global__ void k_group1(const float* __restrict__ xyz, const float* __restrict__ newXyz,
                         const int* __restrict__ idx, half_t* __restrict__ out, long total) {
  long g = (long)blockIdx.x * blockDim.x + threadIdx.x;
  if (g >= total) return;                       // g = (b*S1 + s)*NS + k
  int b = (int)(g / ((long)S1v * NSv));
  int pi = idx[g];
  int center = (int)(g / NSv);
  const float* xin = xyz + (size_t)b * 6 * NPT;
  const float* c = newXyz + (size_t)center * 3;
  half_t* row = out + (size_t)g * 32;
  row[0] = (half_t)(xin[0 * NPT + pi] - c[0]);
  row[1] = (half_t)(xin[1 * NPT + pi] - c[1]);
  row[2] = (half_t)(xin[2 * NPT + pi] - c[2]);
  row[3] = (half_t)xin[3 * NPT + pi];
  row[4] = (half_t)xin[4 * NPT + pi];
  row[5] = (half_t)xin[5 * NPT + pi];
  for (int j = 6; j < 32; ++j) row[j] = (half_t)0.0f;
}

__global__ void k_group2(const float* __restrict__ nx1, const half_t* __restrict__ l1pts,
                         const float* __restrict__ nx2, const int* __restrict__ idx,
                         half_t* __restrict__ out, long total) {
  long g = (long)blockIdx.x * blockDim.x + threadIdx.x;
  if (g >= total) return;                       // g = (b*S2 + s)*NS + k
  int b = (int)(g / ((long)S2v * NSv));
  int pi = idx[g];
  int center = (int)(g / NSv);
  const float* src = nx1 + (size_t)(b * S1v + pi) * 3;
  const float* c = nx2 + (size_t)center * 3;
  const half_t* f = l1pts + (size_t)(b * S1v + pi) * 128;
  half_t* row = out + (size_t)g * 160;
  row[0] = (half_t)(src[0] - c[0]);
  row[1] = (half_t)(src[1] - c[1]);
  row[2] = (half_t)(src[2] - c[2]);
  for (int j = 0; j < 128; ++j) row[3 + j] = f[j];
  for (int j = 131; j < 160; ++j) row[j] = (half_t)0.0f;
}

__global__ void k_group3(const float* __restrict__ nx2, const half_t* __restrict__ l2pts,
                         half_t* __restrict__ out, int total) {
  int t = blockIdx.x * blockDim.x + threadIdx.x;
  if (t >= total) return;                       // t = b*S2 + s
  const float* c = nx2 + (size_t)t * 3;
  const half_t* f = l2pts + (size_t)t * 256;
  half_t* row = out + (size_t)t * 288;
  row[0] = (half_t)c[0]; row[1] = (half_t)c[1]; row[2] = (half_t)c[2];
  for (int j = 0; j < 256; ++j) row[3 + j] = f[j];
  for (int j = 259; j < 288; ++j) row[j] = (half_t)0.0f;
}

// ---------------------------------------------------------------------------
// WMMA GEMM: out[p][o] = sum_c act[p][c] * W[o][c] + bias[o]   (f16 in, f32 acc)
// One wave = one 16x16 (points x out-ch) tile; K-loop step 32.
// ---------------------------------------------------------------------------
__global__ __launch_bounds__(256)
void k_gemm(const half_t* __restrict__ act, const half_t* __restrict__ w,
            const float* __restrict__ bias, half_t* __restrict__ out,
            int P, int OUTC, int CINPAD, int biasN) {
  int wave = threadIdx.x >> 5;
  int lane = threadIdx.x & 31;
  int ptile = blockIdx.x * WPB + wave;
  if (ptile * 16 >= P) return;                 // uniform per-wave: EXEC stays full
  int pbase = ptile * 16;
  int obase = blockIdx.y * 16;
  int hs  = lane >> 4;                         // half-wave select
  int l15 = lane & 15;

  // A fragment: row m = obase+l15, K runs {kbase..+7, kbase+16..+23}, kbase = hs*8
  const half_t* aptr = w + (size_t)(obase + l15) * CINPAD + hs * 8;
  // B fragment: col n -> point pbase+l15, K runs {kbase..+7, kbase+16..+23}, kbase = hs*8
  const half_t* bptr = act + (size_t)(pbase + l15) * CINPAD + hs * 8;

  v8f acc = {};
  for (int kc = 0; kc < CINPAD; kc += 32) {
    v8h a0 = *reinterpret_cast<const v8h*>(aptr + kc);
    v8h a1 = *reinterpret_cast<const v8h*>(aptr + kc + 16);
    v8h b0 = *reinterpret_cast<const v8h*>(bptr + kc);
    v8h b1 = *reinterpret_cast<const v8h*>(bptr + kc + 16);
    v16h fa, fb;
#pragma unroll
    for (int i = 0; i < 8; ++i) { fa[i] = a0[i]; fa[i + 8] = a1[i]; fb[i] = b0[i]; fb[i + 8] = b1[i]; }
    acc = __builtin_amdgcn_wmma_f32_16x16x32_f16(false, fa, false, fb,
                                                 (short)0, acc, false, false);
  }

  // D layout: lane owns point pbase+l15, 8 contiguous out channels at obase+hs*8
  int o0 = obase + hs * 8;
  v8h res;
#pragma unroll
  for (int r = 0; r < 8; ++r) {
    int o = o0 + r;
    float v = acc[r] + (o < biasN ? bias[o] : 0.0f);
    res[r] = (half_t)v;
  }
  *reinterpret_cast<v8h*>(out + (size_t)(pbase + l15) * OUTC + o0) = res;
}

// ---------------------------------------------------------------------------
// BN stats (deterministic two-stage) + normalize/ReLU
// ---------------------------------------------------------------------------
__global__ __launch_bounds__(256)
void k_stats(const half_t* __restrict__ x, float* __restrict__ ps, float* __restrict__ pq,
             int P, int C) {
  int c = blockIdx.x, chunk = blockIdx.y;
  __shared__ float s1[256], s2[256];
  float a = 0.f, b = 0.f;
  for (int p = chunk * 256 + threadIdx.x; p < P; p += NCHUNK * 256) {
    float v = (float)x[(size_t)p * C + c];
    a += v; b += v * v;
  }
  s1[threadIdx.x] = a; s2[threadIdx.x] = b;
  __syncthreads();
  for (int s = 128; s > 0; s >>= 1) {
    if (threadIdx.x < s) { s1[threadIdx.x] += s1[threadIdx.x + s]; s2[threadIdx.x] += s2[threadIdx.x + s]; }
    __syncthreads();
  }
  if (threadIdx.x == 0) { ps[c * NCHUNK + chunk] = s1[0]; pq[c * NCHUNK + chunk] = s2[0]; }
}

__global__ void k_stats2(const float* __restrict__ ps, const float* __restrict__ pq,
                         float* __restrict__ sum, float* __restrict__ sq, int C) {
  int c = blockIdx.x * blockDim.x + threadIdx.x;
  if (c >= C) return;
  float a = 0.f, b = 0.f;
  for (int i = 0; i < NCHUNK; ++i) { a += ps[c * NCHUNK + i]; b += pq[c * NCHUNK + i]; }
  sum[c] = a; sq[c] = b;
}

__global__ void k_bnrelu(half_t* __restrict__ x, const float* __restrict__ sum,
                         const float* __restrict__ sq, const float* __restrict__ gamma,
                         const float* __restrict__ beta, int P, int C, long total) {
  long i = (long)blockIdx.x * blockDim.x + threadIdx.x;
  if (i >= total) return;
  int c = (int)(i % C);
  float invP = 1.0f / (float)P;
  float mu = sum[c] * invP;
  float var = sq[c] * invP - mu * mu;
  float v = (float)x[i];
  v = (v - mu) * rsqrtf(var + 1e-5f) * gamma[c] + beta[c];
  x[i] = (half_t)fmaxf(v, 0.0f);
}

// ---------------------------------------------------------------------------
// Max-pool over K samples per group
// ---------------------------------------------------------------------------
__global__ void k_maxpool(const half_t* __restrict__ in, half_t* __restrict__ out,
                          int G, int K, int C) {
  long t = (long)blockIdx.x * blockDim.x + threadIdx.x;
  if (t >= (long)G * C) return;
  int g = (int)(t / C), c = (int)(t % C);
  const half_t* p = in + (size_t)g * K * C + c;
  float m = -1e30f;
  for (int k = 0; k < K; ++k) m = fmaxf(m, (float)p[(size_t)k * C]);
  out[(size_t)g * C + c] = (half_t)m;
}

// ---------------------------------------------------------------------------
// Output writers (tuple order: x, l2_pts^T, l3_pts)
// ---------------------------------------------------------------------------
__global__ void k_out_x(const half_t* __restrict__ fc3, float* __restrict__ out) {
  int t = blockIdx.x * blockDim.x + threadIdx.x;
  if (t >= NB * 63) return;
  int b = t / 63, j = t % 63;
  out[t] = (float)fc3[b * 64 + j];
}

__global__ void k_out_l2(const half_t* __restrict__ pooled2, float* __restrict__ out) {
  long t = (long)blockIdx.x * blockDim.x + threadIdx.x;
  if (t >= (long)NB * 256 * S2v) return;       // (b, c, s) order
  int b = (int)(t / (256 * S2v));
  int r = (int)(t % (256 * S2v));
  int c = r / S2v, s = r % S2v;
  out[t] = (float)pooled2[((size_t)(b * S2v + s)) * 256 + c];
}

__global__ void k_out_l3(const half_t* __restrict__ l3, float* __restrict__ out) {
  int t = blockIdx.x * blockDim.x + threadIdx.x;
  if (t >= NB * 1024) return;
  out[t] = (float)l3[t];
}

// ---------------------------------------------------------------------------
// Host orchestration
// ---------------------------------------------------------------------------
struct LDef { int widx; int cin; int cout; int bn; };
static inline int rnd32(int x) { return (x + 31) & ~31; }

extern "C" void kernel_launch(void* const* d_in, const int* in_sizes, int n_in,
                              void* d_out, int out_size, void* d_ws, size_t ws_size,
                              hipStream_t stream) {
  (void)in_sizes; (void)n_in; (void)out_size;
  const float* xyz = (const float*)d_in[0];
  float* dout = (float*)d_out;

  // Param order: insertion order of setup_inputs(): sa1[3], sa2[3], sa3[3], fc1, fc2, fc3;
  // each layer contributes w, b, gamma, beta (fc3: w, b).
  static const LDef L[12] = {
    {1, 6, 64, 1},   {5, 64, 64, 1},   {9, 64, 128, 1},
    {13, 131, 128, 1},{17, 128, 128, 1},{21, 128, 256, 1},
    {25, 259, 256, 1},{29, 256, 256, 1},{33, 256, 1024, 1},
    {37, 1024, 512, 1},{41, 512, 128, 1},{45, 128, 63, 0},
  };

  // ---- workspace bump allocator ----
  char* ws = (char*)d_ws;
  size_t off = 0;
  auto alloc = [&](size_t bytes) -> void* {
    void* p = ws + off;
    off = (off + bytes + 255) & ~(size_t)255;
    return p;
  };

  half_t* wbuf[12];
  for (int i = 0; i < 12; ++i)
    wbuf[i] = (half_t*)alloc((size_t)rnd32(L[i].cout) * rnd32(L[i].cin) * sizeof(half_t));

  int*   fps1  = (int*)  alloc((size_t)NB * S1v * sizeof(int));
  float* nx1   = (float*)alloc((size_t)NB * S1v * 3 * sizeof(float));
  int*   fps2  = (int*)  alloc((size_t)NB * S2v * sizeof(int));
  float* nx2   = (float*)alloc((size_t)NB * S2v * 3 * sizeof(float));
  int*   ball1 = (int*)  alloc((size_t)NB * S1v * NSv * sizeof(int));
  int*   ball2 = (int*)  alloc((size_t)NB * S2v * NSv * sizeof(int));
  half_t* l1pts = (half_t*)alloc((size_t)NB * S1v * 128 * sizeof(half_t));
  half_t* l2pts = (half_t*)alloc((size_t)NB * S2v * 256 * sizeof(half_t));
  half_t* l3pts = (half_t*)alloc((size_t)NB * 1024 * sizeof(half_t));
  float* psum = (float*)alloc((size_t)1024 * NCHUNK * sizeof(float));
  float* psq  = (float*)alloc((size_t)1024 * NCHUNK * sizeof(float));
  float* csum = (float*)alloc(1024 * sizeof(float));
  float* csq  = (float*)alloc(1024 * sizeof(float));
  half_t* bufA = (half_t*)alloc((size_t)134217728);   // <= 1M pts x 64ch f16
  half_t* bufB = (half_t*)alloc((size_t)268435456);   // <= 1M pts x 128ch f16
  if (off > ws_size) return;                          // insufficient scratch

  // ---- convert + pad all weights to f16 (deterministic, every call) ----
  for (int i = 0; i < 12; ++i) {
    long tot = (long)rnd32(L[i].cout) * rnd32(L[i].cin);
    k_convert_w<<<(unsigned)((tot + 255) / 256), 256, 0, stream>>>(
        (const float*)d_in[L[i].widx], wbuf[i], L[i].cout, L[i].cin, rnd32(L[i].cin), tot);
  }

  auto run_layer = [&](const half_t* in, half_t* out, int P, int li) {
    int cinPad = rnd32(L[li].cin), coutPad = rnd32(L[li].cout);
    dim3 g((P + 16 * WPB - 1) / (16 * WPB), coutPad / 16);
    k_gemm<<<g, 256, 0, stream>>>(in, wbuf[li], (const float*)d_in[L[li].widx + 1],
                                  out, P, coutPad, cinPad, L[li].cout);
    if (L[li].bn) {
      k_stats<<<dim3(coutPad, NCHUNK), 256, 0, stream>>>(out, psum, psq, P, coutPad);
      k_stats2<<<(coutPad + 255) / 256, 256, 0, stream>>>(psum, psq, csum, csq, coutPad);
      long total = (long)P * coutPad;
      k_bnrelu<<<(unsigned)((total + 255) / 256), 256, 0, stream>>>(
          out, csum, csq, (const float*)d_in[L[li].widx + 2],
          (const float*)d_in[L[li].widx + 3], P, coutPad, total);
    }
  };

  // ================= SA1 =================
  k_fps<<<NB, 256, 0, stream>>>(xyz, 6L * NPT, 1L, (long)NPT, NPT, S1v, fps1, nx1);
  k_ball<<<(NB * S1v + 255) / 256, 256, 0, stream>>>(xyz, 6L * NPT, 1L, (long)NPT,
                                                     NPT, nx1, S1v, 0.015f * 0.015f,
                                                     ball1, NB * S1v);
  long P1 = (long)NB * S1v * NSv;               // 1,048,576
  k_group1<<<(unsigned)((P1 + 255) / 256), 256, 0, stream>>>(xyz, nx1, ball1, bufA, P1);
  run_layer(bufA, bufB, (int)P1, 0);            // 32 -> 64
  run_layer(bufB, bufA, (int)P1, 1);            // 64 -> 64
  run_layer(bufA, bufB, (int)P1, 2);            // 64 -> 128
  k_maxpool<<<(unsigned)(((long)NB * S1v * 128 + 255) / 256), 256, 0, stream>>>(
      bufB, l1pts, NB * S1v, NSv, 128);

  // ================= SA2 =================
  k_fps<<<NB, 256, 0, stream>>>(nx1, (long)S1v * 3, 3L, 1L, S1v, S2v, fps2, nx2);
  k_ball<<<(NB * S2v + 255) / 256, 256, 0, stream>>>(nx1, (long)S1v * 3, 3L, 1L,
                                                     S1v, nx2, S2v, 0.04f * 0.04f,
                                                     ball2, NB * S2v);
  long P2 = (long)NB * S2v * NSv;               // 262,144
  k_group2<<<(unsigned)((P2 + 255) / 256), 256, 0, stream>>>(nx1, l1pts, nx2, ball2, bufA, P2);
  run_layer(bufA, bufB, (int)P2, 3);            // 160 -> 128
  run_layer(bufB, bufA, (int)P2, 4);            // 128 -> 128
  run_layer(bufA, bufB, (int)P2, 5);            // 128 -> 256
  k_maxpool<<<(unsigned)(((long)NB * S2v * 256 + 255) / 256), 256, 0, stream>>>(
      bufB, l2pts, NB * S2v, NSv, 256);
  k_out_l2<<<(unsigned)(((long)NB * 256 * S2v + 255) / 256), 256, 0, stream>>>(
      l2pts, dout + 2016);

  // ================= SA3 (global) =================
  int P3 = NB * S2v;                             // 4096
  k_group3<<<(P3 + 255) / 256, 256, 0, stream>>>(nx2, l2pts, bufA, P3);
  run_layer(bufA, bufB, P3, 6);                 // 288 -> 256
  run_layer(bufB, bufA, P3, 7);                 // 256 -> 256
  run_layer(bufA, bufB, P3, 8);                 // 256 -> 1024
  k_maxpool<<<(NB * 1024 + 255) / 256, 256, 0, stream>>>(bufB, l3pts, NB, S2v, 1024);
  k_out_l3<<<(NB * 1024 + 255) / 256, 256, 0, stream>>>(l3pts, dout + 2016 + NB * 256 * S2v);

  // ================= FC head =================
  run_layer(l3pts, bufA, NB, 9);                // 1024 -> 512, BN+ReLU over batch
  run_layer(bufA, bufB, NB, 10);                // 512 -> 128
  run_layer(bufB, bufA, NB, 11);                // 128 -> 64(pad), no BN
  k_out_x<<<(NB * 63 + 255) / 256, 256, 0, stream>>>(bufA, dout);
}
